// QKNormedAttention_57303453663802
// MI455X (gfx1250) — compile-verified, matrix operand order checked by reading, not run on hardware
//
#include <hip/hip_runtime.h>
#include <hip/hip_bf16.h>

// ---------------------------------------------------------------------------
// QKNormedAttention for MI455X (gfx1250), wave32, WMMA f16->f32,
// async global->LDS tile staging (ASYNCcnt), double-buffered.
// N=2048 tokens, DIM=1024, H=16 heads, HD=64.
// ---------------------------------------------------------------------------

typedef __attribute__((ext_vector_type(4)))  _Float16 v4h;
typedef __attribute__((ext_vector_type(8)))  _Float16 v8h;
typedef __attribute__((ext_vector_type(16))) _Float16 v16h;
typedef __attribute__((ext_vector_type(8)))  float    v8f;

#define N_TOK 2048
#define DIM   1024
#define HEADS 16
#define HD    64

static __device__ __forceinline__ v16h cat16(v8h lo, v8h hi) {
  return __builtin_shufflevector(lo, hi, 0,1,2,3,4,5,6,7,8,9,10,11,12,13,14,15);
}

// Flat LDS addresses carry the LDS byte offset in addr[31:0] (ISA 10.2).
static __device__ __forceinline__ unsigned lds_off(const void* p) {
  return (unsigned)(uintptr_t)p;
}
// CDNA5 async global->LDS copy, bypasses VGPRs, tracked by ASYNCcnt.
static __device__ __forceinline__ void async_ld_b128(unsigned lds, const void* g) {
  asm volatile("global_load_async_to_lds_b128 %0, %1, off"
               :: "v"(lds), "v"(g) : "memory");
}
static __device__ __forceinline__ void wait_async0() {
  asm volatile("s_wait_asynccnt 0x0" ::: "memory");
}

// ------------------------------ fp32 -> f16 --------------------------------
__global__ __launch_bounds__(256) void cvt_f32_f16(const float* __restrict__ in,
                                                   _Float16* __restrict__ out, int n4) {
  int i = blockIdx.x * blockDim.x + threadIdx.x;
  int stride = gridDim.x * blockDim.x;
  const float4* in4 = (const float4*)in;
  v4h* out4 = (v4h*)out;
  for (; i < n4; i += stride) {
    float4 v = in4[i];
    v4h o;
    o[0] = (_Float16)v.x; o[1] = (_Float16)v.y;
    o[2] = (_Float16)v.z; o[3] = (_Float16)v.w;
    out4[i] = o;
  }
}

// ------------------- RMSNorm over DIM, write scaled f16 --------------------
__global__ __launch_bounds__(256) void rmsnorm_f16(const float* __restrict__ x,
                                                   const float* __restrict__ w,
                                                   const float* __restrict__ b,
                                                   _Float16* __restrict__ out,
                                                   float outscale) {
  const unsigned row = blockIdx.x;
  const unsigned t = threadIdx.x;            // 256 threads * 4 floats = 1024
  const float4* xr = (const float4*)(x + row * DIM);
  float4 v = xr[t];
  float ss = v.x * v.x + v.y * v.y + v.z * v.z + v.w * v.w;
  #pragma unroll
  for (int off = 16; off > 0; off >>= 1) ss += __shfl_xor(ss, off, 32);
  __shared__ float wsum[8];
  if ((t & 31) == 0) wsum[t >> 5] = ss;
  __syncthreads();
  float tot = 0.f;
  #pragma unroll
  for (int i = 0; i < 8; i++) tot += wsum[i];
  float inv = rsqrtf(tot * (1.0f / DIM) + 1e-5f) * outscale;
  float4 wv = ((const float4*)w)[t];
  float4 bv = ((const float4*)b)[t];
  v4h o;
  o[0] = (_Float16)((v.x * inv) * wv.x + bv.x * outscale);
  o[1] = (_Float16)((v.y * inv) * wv.y + bv.y * outscale);
  o[2] = (_Float16)((v.z * inv) * wv.z + bv.z * outscale);
  o[3] = (_Float16)((v.w * inv) * wv.w + bv.w * outscale);
  ((v4h*)(out + row * DIM))[t] = o;
}

// -------------- transpose v_f32 [N,DIM] -> vt16 [DIM rows][N] --------------
__global__ void transpose_vt(const float* __restrict__ v, _Float16* __restrict__ vt) {
  __shared__ float tile[32][33];
  const unsigned m0 = blockIdx.x * 32;
  const unsigned d0 = blockIdx.y * 32;
  const unsigned tx = threadIdx.x, ty = threadIdx.y;   // 32 x 8
  #pragma unroll
  for (int i = 0; i < 32; i += 8)
    tile[ty + i][tx] = v[(m0 + ty + i) * DIM + d0 + tx];
  __syncthreads();
  #pragma unroll
  for (int i = 0; i < 32; i += 8)
    vt[(d0 + ty + i) * N_TOK + m0 + tx] = (_Float16)tile[tx][ty + i];
}

// --------------------- NT GEMM: C = A(MxK) * B(NxK)^T + bias ---------------
// BM=128 BN=64 BK=32; 8 waves, each 32x32 (2x2 WMMA tiles).
// Async global->LDS staging, double buffered.
#define LDK 40   // padded LDS K-stride (halves): 80B rows, b128-aligned
__global__ __launch_bounds__(256) void gemm_nt_wmma(const _Float16* __restrict__ A,
                                                    const _Float16* __restrict__ B,
                                                    const float* __restrict__ bias,
                                                    float* __restrict__ C,
                                                    int M, int N, int K) {
  __shared__ __align__(16) _Float16 As[2][128 * LDK];
  __shared__ __align__(16) _Float16 Bs[2][64 * LDK];
  const unsigned tid = threadIdx.x;
  const unsigned m0 = blockIdx.y * 128;
  const unsigned n0 = blockIdx.x * 64;
  const unsigned wave = tid >> 5;
  const unsigned lane = tid & 31;
  const unsigned wm = wave >> 1;   // 0..3 -> m sub-block * 32
  const unsigned wn = wave & 1;    // 0..1 -> n sub-block * 32
  const unsigned lh = lane >> 4, lr = lane & 15;
  const unsigned uK = (unsigned)K, uN = (unsigned)N;

  auto loadAB = [&](int buf, unsigned k0) {
    #pragma unroll
    for (int q2 = 0; q2 < 2; q2++) {         // A: 512 b128 chunks, 2/thread
      unsigned idx = tid + q2 * 256;
      unsigned r = idx >> 2, c8 = (idx & 3) * 8;
      async_ld_b128(lds_off(&As[buf][r * LDK + c8]),
                    &A[(m0 + r) * uK + k0 + c8]);
    }
    {                                         // B: 256 chunks, 1/thread
      unsigned r = tid >> 2, c8 = (tid & 3) * 8;
      async_ld_b128(lds_off(&Bs[buf][r * LDK + c8]),
                    &B[(n0 + r) * uK + k0 + c8]);
    }
  };

  v8f c[2][2] = {};
  loadAB(0, 0);
  int cur = 0;
  for (unsigned k0 = 0; k0 < uK; k0 += 32) {
    wait_async0();                 // our async stores into buf[cur] landed
    __syncthreads();               // everyone's landed; prior reads finished
    if (k0 + 32 < uK) loadAB(cur ^ 1, k0 + 32);   // stream next tile

    v16h a[2], b[2];
    #pragma unroll
    for (int t = 0; t < 2; t++) {
      unsigned row = wm * 32 + t * 16 + lr;
      a[t] = cat16(*(const v8h*)&As[cur][row * LDK + lh * 8],
                   *(const v8h*)&As[cur][row * LDK + 16 + lh * 8]);
    }
    #pragma unroll
    for (int t = 0; t < 2; t++) {
      unsigned row = wn * 32 + t * 16 + lr;
      b[t] = cat16(*(const v8h*)&Bs[cur][row * LDK + lh * 8],
                   *(const v8h*)&Bs[cur][row * LDK + 16 + lh * 8]);
    }
    #pragma unroll
    for (int i = 0; i < 2; i++)
      #pragma unroll
      for (int j = 0; j < 2; j++)
        c[i][j] = __builtin_amdgcn_wmma_f32_16x16x32_f16(
            false, a[i], false, b[j], (short)0, c[i][j], false, false);
    cur ^= 1;
  }
  // epilogue: C layout — VGPR r: row = r + 8*laneHalf, col = lane%16
  #pragma unroll
  for (int i = 0; i < 2; i++)
    #pragma unroll
    for (int j = 0; j < 2; j++) {
      unsigned col = n0 + wn * 32 + j * 16 + lr;
      float bb = bias[col];
      unsigned base = (m0 + wm * 32 + i * 16 + lh * 8) * uN + col;
      #pragma unroll
      for (int r = 0; r < 8; r++)
        C[base + r * uN] = c[i][j][r] + bb;
    }
}

// --------------------------- flash attention -------------------------------
// grid(16 qblocks, 16 heads), 8 waves, wave = 16 queries.
// S^T = K*Q^T so softmax keys land on the VGPR axis (one xor-16 shuffle).
// K/V tiles staged once per block via async->LDS, double buffered.
#define KLD 72   // K tile LDS stride (halves): 144B rows
#define VLD 40   // V tile LDS stride (halves): 80B rows
__global__ __launch_bounds__(256) void attn_wmma(const _Float16* __restrict__ q16,
                                                 const _Float16* __restrict__ k16,
                                                 const _Float16* __restrict__ vt16,
                                                 _Float16* __restrict__ o16) {
  __shared__ __align__(16) _Float16 Ks[2][32 * KLD];  // 32 keys x 64 d
  __shared__ __align__(16) _Float16 Vs[2][64 * VLD];  // 64 d x 32 keys
  const unsigned h = blockIdx.y;
  const unsigned wave = threadIdx.x >> 5, lane = threadIdx.x & 31;
  const unsigned lh = lane >> 4, lr = lane & 15;
  const unsigned qrow = blockIdx.x * 128 + wave * 16 + lr;

  auto loadKV = [&](int buf, unsigned j) {
    unsigned t = threadIdx.x;
    { // K tile: 32 rows x 8 chunks
      unsigned r = t >> 3, c8 = (t & 7) * 8;
      async_ld_b128(lds_off(&Ks[buf][r * KLD + c8]),
                    &k16[(j + r) * DIM + h * HD + c8]);
    }
    { // V^T tile: 64 rows x 4 chunks
      unsigned r = t >> 2, c8 = (t & 3) * 8;
      async_ld_b128(lds_off(&Vs[buf][r * VLD + c8]),
                    &vt16[(h * HD + r) * N_TOK + j + c8]);
    }
  };

  // Q^T B-fragments (persistent): K = head-dim 0..63 -> 2 frags
  const _Float16* qb = q16 + qrow * DIM + h * HD;
  v16h qf[2];
  #pragma unroll
  for (int t = 0; t < 2; t++)
    qf[t] = cat16(*(const v8h*)&qb[t * 32 + lh * 8],
                  *(const v8h*)&qb[t * 32 + 16 + lh * 8]);

  v8f o[4] = {};                 // O^T accumulators (4 d-tiles x 16 queries)
  float mrun = -1e30f, lsum = 0.f;

  loadKV(0, 0);
  int cur = 0;
  for (unsigned j = 0; j < N_TOK; j += 32) {
    wait_async0();
    __syncthreads();
    if (j + 32 < N_TOK) loadKV(cur ^ 1, j + 32);

    // S^T tiles: rows = keys, cols = queries; logits already in log2-domain
    v8f s[2];
    #pragma unroll
    for (int st = 0; st < 2; st++) {
      const _Float16* kb = &Ks[cur][(st * 16 + lr) * KLD];
      v16h a0 = cat16(*(const v8h*)&kb[lh * 8],      *(const v8h*)&kb[16 + lh * 8]);
      v16h a1 = cat16(*(const v8h*)&kb[32 + lh * 8], *(const v8h*)&kb[48 + lh * 8]);
      v8f z = {};
      z = __builtin_amdgcn_wmma_f32_16x16x32_f16(false, a0, false, qf[0], (short)0, z, false, false);
      z = __builtin_amdgcn_wmma_f32_16x16x32_f16(false, a1, false, qf[1], (short)0, z, false, false);
      s[st] = z;
    }
    // online softmax (base-2; 1/sqrt(hd)*log2e folded into q16)
    float bm = -1e30f;
    #pragma unroll
    for (int st = 0; st < 2; st++)
      #pragma unroll
      for (int r = 0; r < 8; r++) bm = fmaxf(bm, s[st][r]);
    bm = fmaxf(bm, __shfl_xor(bm, 16, 32));
    float mnew = fmaxf(mrun, bm);
    float alpha = exp2f(mrun - mnew);
    float psum = 0.f;
    v16h pb;                     // P^T tiles pack directly into a B fragment
    #pragma unroll
    for (int st = 0; st < 2; st++)
      #pragma unroll
      for (int r = 0; r < 8; r++) {
        float p = exp2f(s[st][r] - mnew);
        psum += p;
        pb[st * 8 + r] = (_Float16)p;
      }
    psum += __shfl_xor(psum, 16, 32);
    lsum = lsum * alpha + psum;
    mrun = mnew;
    #pragma unroll
    for (int dt = 0; dt < 4; dt++)
      #pragma unroll
      for (int r = 0; r < 8; r++) o[dt][r] *= alpha;
    // O^T += V^T * P^T
    #pragma unroll
    for (int dt = 0; dt < 4; dt++) {
      const _Float16* vb = &Vs[cur][(dt * 16 + lr) * VLD];
      v16h va = cat16(*(const v8h*)&vb[lh * 8], *(const v8h*)&vb[16 + lh * 8]);
      o[dt] = __builtin_amdgcn_wmma_f32_16x16x32_f16(false, va, false, pb, (short)0, o[dt], false, false);
    }
    cur ^= 1;
  }
  float inv = 1.0f / lsum;
  _Float16* ob = o16 + qrow * DIM + h * HD;
  #pragma unroll
  for (int dt = 0; dt < 4; dt++) {
    v8h pk;
    #pragma unroll
    for (int r = 0; r < 8; r++) pk[r] = (_Float16)(o[dt][r] * inv);
    *(v8h*)&ob[dt * 16 + lh * 8] = pk;     // d contiguous: one b128 store
  }
}

// ------------------------------- launcher ----------------------------------
extern "C" void kernel_launch(void* const* d_in, const int* in_sizes, int n_in,
                              void* d_out, int out_size, void* d_ws, size_t ws_size,
                              hipStream_t stream) {
  (void)in_sizes; (void)n_in; (void)out_size; (void)ws_size;
  const float* query = (const float*)d_in[0];
  const float* keyt  = (const float*)d_in[1];
  const float* value = (const float*)d_in[2];
  const float* Wq = (const float*)d_in[3];  const float* bq = (const float*)d_in[4];
  const float* Wk = (const float*)d_in[5];  const float* bk = (const float*)d_in[6];
  const float* Wv = (const float*)d_in[7];  const float* bv = (const float*)d_in[8];
  const float* Wo = (const float*)d_in[9];  const float* bo = (const float*)d_in[10];
  const float* qn_w = (const float*)d_in[11]; const float* qn_b = (const float*)d_in[12];
  const float* kn_w = (const float*)d_in[13]; const float* kn_b = (const float*)d_in[14];
  float* out = (float*)d_out;

  // carve workspace (256B aligned)
  size_t off = 0;
  auto carve = [&](size_t bytes) {
    void* p = (char*)d_ws + off;
    off += (bytes + 255) & ~(size_t)255;
    return p;
  };
  const size_t ND = (size_t)N_TOK * DIM, DD = (size_t)DIM * DIM;
  _Float16* q_in16 = (_Float16*)carve(ND * 2);
  _Float16* k_in16 = (_Float16*)carve(ND * 2);
  _Float16* v_in16 = (_Float16*)carve(ND * 2);
  _Float16* Wq16   = (_Float16*)carve(DD * 2);
  _Float16* Wk16   = (_Float16*)carve(DD * 2);
  _Float16* Wv16   = (_Float16*)carve(DD * 2);
  _Float16* Wo16   = (_Float16*)carve(DD * 2);
  float*    qf     = (float*)   carve(ND * 4);
  float*    kf     = (float*)   carve(ND * 4);
  float*    vf     = (float*)   carve(ND * 4);
  _Float16* q16n   = (_Float16*)carve(ND * 2);
  _Float16* k16n   = (_Float16*)carve(ND * 2);
  _Float16* vt16   = (_Float16*)carve(ND * 2);
  _Float16* attn16 = (_Float16*)carve(ND * 2);

  // 1) convert inputs + weights to f16
  cvt_f32_f16<<<2048, 256, 0, stream>>>(query, q_in16, (int)(ND / 4));
  cvt_f32_f16<<<2048, 256, 0, stream>>>(keyt,  k_in16, (int)(ND / 4));
  cvt_f32_f16<<<2048, 256, 0, stream>>>(value, v_in16, (int)(ND / 4));
  cvt_f32_f16<<<1024, 256, 0, stream>>>(Wq, Wq16, (int)(DD / 4));
  cvt_f32_f16<<<1024, 256, 0, stream>>>(Wk, Wk16, (int)(DD / 4));
  cvt_f32_f16<<<1024, 256, 0, stream>>>(Wv, Wv16, (int)(DD / 4));
  cvt_f32_f16<<<1024, 256, 0, stream>>>(Wo, Wo16, (int)(DD / 4));

  // 2) projections: x @ W^T + b  (fp32 accumulate)
  dim3 ggrid(DIM / 64, N_TOK / 128);
  gemm_nt_wmma<<<ggrid, 256, 0, stream>>>(q_in16, Wq16, bq, qf, N_TOK, DIM, DIM);
  gemm_nt_wmma<<<ggrid, 256, 0, stream>>>(k_in16, Wk16, bk, kf, N_TOK, DIM, DIM);
  gemm_nt_wmma<<<ggrid, 256, 0, stream>>>(v_in16, Wv16, bv, vf, N_TOK, DIM, DIM);

  // 3) RMSNorm; fold softmax scale (1/sqrt(64)=0.125) and log2(e) into q
  const float qscale = 0.125f * 1.44269504088896340736f;
  rmsnorm_f16<<<N_TOK, 256, 0, stream>>>(qf, qn_w, qn_b, q16n, qscale);
  rmsnorm_f16<<<N_TOK, 256, 0, stream>>>(kf, kn_w, kn_b, k16n, 1.0f);

  // 4) V transpose -> [h][d][m] f16 for the P*V WMMA A-operand
  transpose_vt<<<dim3(N_TOK / 32, DIM / 32), dim3(32, 8), 0, stream>>>(vf, vt16);

  // 5) attention
  attn_wmma<<<dim3(N_TOK / 128, HEADS), 256, 0, stream>>>(q16n, k16n, vt16, attn16);

  // 6) output projection: attn @ Wo^T + bo -> fp32 out
  gemm_nt_wmma<<<ggrid, 256, 0, stream>>>(attn16, Wo16, bo, out, N_TOK, DIM, DIM);
}
